// AttMultiHeadMultiResLoc_65549790871852
// MI455X (gfx1250) — matrix operand order repaired and checked by reading, not compiled, not run on hardware
//
#include <hip/hip_runtime.h>
#include <hip/hip_bf16.h>

// ---------------------------------------------------------------------------
// Shapes (compile-time constants from the reference)
// ---------------------------------------------------------------------------
#define H_    4
#define B_    32
#define T_    1000
#define EPROJS_ 1024
#define DUNITS_ 1024
#define DK_   512
#define DV_   512
#define BT_   (B_ * T_)          // 32000 rows, divisible by 32
#define WTK_  1248               // 1024 + 224 (max padded conv taps); rows 2496B (32B aligned)
#define APL_  1224               // padded att_prev row length: 1000 + 224
#define UCH_  8                  // T-chunks for the weighted-sum kernel (125 each)

typedef __attribute__((ext_vector_type(16))) __bf16 v16bf;
typedef __attribute__((ext_vector_type(8)))  float  v8f;

#if defined(__has_builtin)
#if __has_builtin(__builtin_amdgcn_global_load_async_to_lds_b64)
#define ASYNC_LDS 1
#endif
#endif

#if defined(ASYNC_LDS)
// exact parameter types from the hipcc diagnostic: v2i with HIP __device__ (AS1)
// for the global source; LDS side is the AS3 twin.
typedef int ai2 __attribute__((vector_size(2 * sizeof(int))));
typedef __attribute__((address_space(1))) ai2 ga2;   // global (prints as __device__)
typedef __attribute__((address_space(3))) ai2 la2;   // LDS    (prints as __shared__)
#endif

__device__ __forceinline__ unsigned short f2bf(float f) {
  unsigned int u = __float_as_uint(f);
  u += 0x7FFFu + ((u >> 16) & 1u);     // round-to-nearest-even
  return (unsigned short)(u >> 16);
}

// padded conv-tap count per head: K_h = 50(h+1)+1 -> {51,101,151,201} -> {64,128,160,224}
__device__ __forceinline__ int kpad_of(int h) { return ((50 * (h + 1) + 32) >> 5) << 5; }

// ---------------------------------------------------------------------------
// 1) enc_hs_pad f32 -> bf16  (32000 x 1024)
// ---------------------------------------------------------------------------
__global__ __launch_bounds__(256) void cvt_x_kernel(const float* __restrict__ x,
                                                    unsigned short* __restrict__ xb) {
  int i = (blockIdx.x * 256 + threadIdx.x) * 4;
  float4 v = *(const float4*)(x + i);
  xb[i + 0] = f2bf(v.x);
  xb[i + 1] = f2bf(v.y);
  xb[i + 2] = f2bf(v.z);
  xb[i + 3] = f2bf(v.w);
}

// ---------------------------------------------------------------------------
// 2) Build transposed augmented weight WT[h][n][k], k in [0,1248)
//      k <  1024          : Wk[h][k][n]
//      k >= 1024, j<K_h   : Weff[j][n] = sum_c conv_w_h[c,j] * Watt[h][c][n]
//      else               : 0
// ---------------------------------------------------------------------------
__global__ __launch_bounds__(256) void build_wt_kernel(const float* __restrict__ Wk,
                                                       const float* __restrict__ Watt,
                                                       const float* __restrict__ cw0,
                                                       const float* __restrict__ cw1,
                                                       const float* __restrict__ cw2,
                                                       const float* __restrict__ cw3,
                                                       unsigned short* __restrict__ WT) {
  int i = blockIdx.x * 256 + threadIdx.x;            // H*512*1248 = 2,555,904 total
  int h = i / (DK_ * WTK_);
  int r = i % (DK_ * WTK_);
  int n = r / WTK_;
  int k = r % WTK_;
  float v = 0.f;
  if (k < EPROJS_) {
    v = Wk[((size_t)h * EPROJS_ + k) * DK_ + n];
  } else {
    int j  = k - EPROJS_;
    int Kh = 50 * (h + 1) + 1;
    if (j < Kh) {
      const float* cw = (h == 0) ? cw0 : (h == 1) ? cw1 : (h == 2) ? cw2 : cw3;
      const float* wa = Watt + (size_t)h * 100 * DK_ + n;
      float a = 0.f;
      for (int c = 0; c < 100; ++c) a += cw[(size_t)c * Kh + j] * wa[(size_t)c * DK_];
      v = a;
    }
  }
  WT[i] = f2bf(v);
}

// ---------------------------------------------------------------------------
// 3) Zero-padded bf16 att_prev: AP[h][b][i] = att_prev[h,b,i-f], f = 25(h+1)
// ---------------------------------------------------------------------------
__global__ __launch_bounds__(256) void build_ap_kernel(const float* __restrict__ att_prev,
                                                       unsigned short* __restrict__ AP) {
  int i = blockIdx.x * 256 + threadIdx.x;            // H*B*1224 = 156,672 total
  int h   = i / (B_ * APL_);
  int r   = i % (B_ * APL_);
  int b   = r / APL_;
  int idx = r % APL_;
  int t = idx - 25 * (h + 1);
  float v = (t >= 0 && t < T_) ? att_prev[((size_t)h * B_ + b) * T_ + t] : 0.f;
  AP[i] = f2bf(v);
}

// ---------------------------------------------------------------------------
// 4) q projection: Qb[h][b][n] = dec_z[b] . Wq[h][:,n] + bq[h][n]
// ---------------------------------------------------------------------------
__global__ __launch_bounds__(256) void qproj_kernel(const float* __restrict__ dec_z,
                                                    const float* __restrict__ Wq,
                                                    const float* __restrict__ bq,
                                                    float* __restrict__ Qb) {
  int i = blockIdx.x * 256 + threadIdx.x;            // H*B*512 = 65536
  int h = i >> 14;
  int b = (i >> 9) & 31;
  int n = i & 511;
  float acc = bq[h * DK_ + n];
  const float* z = dec_z + (size_t)b * DUNITS_;
  const float* w = Wq + (size_t)h * DUNITS_ * DK_ + n;
  for (int d = 0; d < DUNITS_; ++d) acc += z[d] * w[(size_t)d * DK_];
  Qb[i] = acc;
}

// ---------------------------------------------------------------------------
// 5) Fused WMMA kernel (K proj + location conv + q bias + tanh + g_w reduce):
//    E[h][row] = sum_n g_w[h,n]*tanh( (X'@W')[row,n] + q[h,b,n] ) + g_b[h]
//    X' = [ Xbf16 (1024) | att_prev sliding window (kpad(h)) ]
//    grid (1000, H); 8 waves; wave: 2 M-tiles x 4 N-tiles of wmma_f32_16x16x32_bf16
//    LDS A tiles double-buffered, staged with GLOBAL_LOAD_ASYNC_TO_LDS when the
//    toolchain exposes it (drained by s_wait_asynccnt before the barrier).
// ---------------------------------------------------------------------------
__global__ __launch_bounds__(256) void gemm_e_kernel(const unsigned short* __restrict__ Xb,
                                                     const unsigned short* __restrict__ AP,
                                                     const unsigned short* __restrict__ WT,
                                                     const float* __restrict__ Qb,
                                                     const float* __restrict__ g_w,
                                                     const float* __restrict__ g_b,
                                                     float* __restrict__ Eb) {
  const int h    = blockIdx.y;
  const int row0 = blockIdx.x * 32;
  const int wave = threadIdx.x >> 5;
  const int lane = threadIdx.x & 31;
  const int n0   = wave * 64;                        // 4 N-tiles: n0 + {0,16,32,48}
  const int ksteps = 32 + (kpad_of(h) >> 5);         // 34 / 36 / 37 / 39

  __shared__ unsigned short a_s[2][2][16 * 32];      // [buf][mtile][16x32 bf16]
  __shared__ float e_s[8][32];                       // per-wave partial energies

  v8f acc[2][4];
  #pragma unroll
  for (int mt = 0; mt < 2; ++mt)
    #pragma unroll
    for (int j = 0; j < 4; ++j)
      #pragma unroll
      for (int r = 0; r < 8; ++r) acc[mt][j][r] = 0.f;

  const unsigned short* wbase  = WT + (size_t)h * DK_ * WTK_;
  const unsigned short* apbase = AP + (size_t)h * B_ * APL_;

  // per-thread staging geometry: one b64 (4 bf16) per thread, 2KB tile-pair total
  const int srow  = threadIdx.x >> 3;                // 0..31
  const int se    = (threadIdx.x & 7) * 4;           // element within row
  const int sgrow = row0 + srow;
  const int sb    = sgrow / T_;
  const int st    = sgrow % T_;

  auto stage = [&](int ks, int buf) {
    const int kb = ks * 32;
    unsigned short* dst = &a_s[buf][srow >> 4][(srow & 15) * 32 + se];
    if (kb < EPROJS_) {
      const unsigned short* src = Xb + (size_t)sgrow * EPROJS_ + kb + se;
#if defined(ASYNC_LDS)
      __builtin_amdgcn_global_load_async_to_lds_b64((ga2*)src, (la2*)dst, 0, 0);
#else
      *(unsigned long long*)dst = *(const unsigned long long*)src;
#endif
    } else {
      // conv window columns: 2B-aligned gather from padded att_prev
      const unsigned short* ap = apbase + (size_t)sb * APL_ + st + (kb - EPROJS_) + se;
      dst[0] = ap[0]; dst[1] = ap[1]; dst[2] = ap[2]; dst[3] = ap[3];
    }
  };

  stage(0, 0);

  const int am   = lane & 15;                        // A: M = lane (per half)
  const int asel = (lane >> 4) * 8;                  // A: dword group select
  const int ncol = lane & 15;                        // B: column within tile
  const int koff = (lane >> 4) * 16;                 // B: K half select

  for (int ks = 0; ks < ksteps; ++ks) {
#if defined(ASYNC_LDS)
    asm volatile("s_wait_asynccnt 0x0" ::: "memory"); // staged async copies landed
#endif
    __syncthreads();                                 // staged buf[ks&1] now visible
    if (ks + 1 < ksteps) stage(ks + 1, (ks + 1) & 1);

    const int kb = ks * 32;
    v16bf bfrag[4];
    #pragma unroll
    for (int j = 0; j < 4; ++j) {
      const unsigned short* bp = wbase + (size_t)(n0 + j * 16 + ncol) * WTK_ + kb + koff;
      bfrag[j] = *(const v16bf*)bp;
      if (ks + 1 < ksteps) __builtin_prefetch(bp + 32, 0, 3);   // global_prefetch_b8
    }

    #pragma unroll
    for (int mt = 0; mt < 2; ++mt) {
      const unsigned int* ap2 = (const unsigned int*)a_s[ks & 1][mt] + am * 16;
      union { unsigned int u[8]; v16bf v; } af;
      #pragma unroll
      for (int q = 0; q < 4; ++q) { af.u[q] = ap2[asel + q]; af.u[4 + q] = ap2[asel + 4 + q]; }
      #pragma unroll
      for (int j = 0; j < 4; ++j)
        acc[mt][j] = __builtin_amdgcn_wmma_f32_16x16x32_bf16(
            false, af.v, false, bfrag[j], (short)0, acc[mt][j], false, false);
    }
  }

  // epilogue: +q, tanh, *g_w, reduce over N. Each wave writes each of its 32
  // partial rows exactly once (lane0: rows r/16+r, lane16: rows 8+r/24+r),
  // then rows are summed over waves in fixed order -> deterministic.
  const float* gw = g_w + (size_t)h * DK_;
  #pragma unroll
  for (int mt = 0; mt < 2; ++mt) {
    #pragma unroll
    for (int r = 0; r < 8; ++r) {
      int Mrow = mt * 16 + r + (lane >> 4) * 8;      // C layout: VGPR r -> rows r / 8+r
      int grow = row0 + Mrow;
      int b = grow / T_;
      float s = 0.f;
      #pragma unroll
      for (int j = 0; j < 4; ++j) {
        int n = n0 + j * 16 + (lane & 15);
        float val = acc[mt][j][r] + Qb[((size_t)h * B_ + b) * DK_ + n];
        s += tanhf(val) * gw[n];
      }
      s += __shfl_xor(s, 1);
      s += __shfl_xor(s, 2);
      s += __shfl_xor(s, 4);
      s += __shfl_xor(s, 8);
      if ((lane & 15) == 0) e_s[wave][Mrow] = s;
    }
  }
  __syncthreads();
  if (threadIdx.x < 32) {
    float t = 0.f;
    #pragma unroll
    for (int wv = 0; wv < 8; ++wv) t += e_s[wv][threadIdx.x];
    Eb[(size_t)h * BT_ + row0 + threadIdx.x] = t + g_b[h];
  }
}

// ---------------------------------------------------------------------------
// 6) softmax over T with 1/sqrt(DK) scaling; writes w into d_out region
// ---------------------------------------------------------------------------
__global__ __launch_bounds__(256) void softmax_kernel(const float* __restrict__ Eb,
                                                      float* __restrict__ Wout) {
  const int b = blockIdx.x, h = blockIdx.y;
  const float* e = Eb + (size_t)h * BT_ + (size_t)b * T_;
  float* w = Wout + (size_t)h * BT_ + (size_t)b * T_;
  const float scale = 0.04419417382415922f;          // 1/sqrt(512)
  __shared__ float red[256];

  float m = -3.4e38f;
  for (int t = threadIdx.x; t < T_; t += 256) m = fmaxf(m, e[t]);
  red[threadIdx.x] = m;
  __syncthreads();
  for (int s = 128; s > 0; s >>= 1) {
    if (threadIdx.x < s) red[threadIdx.x] = fmaxf(red[threadIdx.x], red[threadIdx.x + s]);
    __syncthreads();
  }
  m = red[0];
  __syncthreads();

  float sum = 0.f;
  for (int t = threadIdx.x; t < T_; t += 256) sum += expf(scale * (e[t] - m));
  red[threadIdx.x] = sum;
  __syncthreads();
  for (int s = 128; s > 0; s >>= 1) {
    if (threadIdx.x < s) red[threadIdx.x] += red[threadIdx.x + s];
    __syncthreads();
  }
  float inv = 1.f / red[0];
  for (int t = threadIdx.x; t < T_; t += 256) w[t] = expf(scale * (e[t] - m)) * inv;
}

// ---------------------------------------------------------------------------
// 7) Upart[ch][h][b][:] = sum_{t in chunk ch} w[h,b,t] * enc[b,t,:]
//    (V projection is linear: the weighted sum commutes with Wv)
// ---------------------------------------------------------------------------
__global__ __launch_bounds__(256) void uacc_kernel(const float* __restrict__ enc,
                                                   const float* __restrict__ w,
                                                   float* __restrict__ Upart) {
  const int b  = blockIdx.x;
  const int ch = blockIdx.y;
  const int t0 = ch * (T_ / UCH_), t1 = t0 + (T_ / UCH_);
  const int d0 = threadIdx.x * 4;
  float acc[H_][4];
  #pragma unroll
  for (int h = 0; h < H_; ++h)
    #pragma unroll
    for (int j = 0; j < 4; ++j) acc[h][j] = 0.f;

  for (int t = t0; t < t1; ++t) {
    float4 x = *(const float4*)(enc + ((size_t)b * T_ + t) * EPROJS_ + d0);
    #pragma unroll
    for (int h = 0; h < H_; ++h) {
      float wv = w[(size_t)h * BT_ + (size_t)b * T_ + t];
      acc[h][0] += wv * x.x; acc[h][1] += wv * x.y;
      acc[h][2] += wv * x.z; acc[h][3] += wv * x.w;
    }
  }
  #pragma unroll
  for (int h = 0; h < H_; ++h)
    #pragma unroll
    for (int j = 0; j < 4; ++j)
      Upart[(((size_t)ch * H_ + h) * B_ + b) * EPROJS_ + d0 + j] = acc[h][j];
}

// fixed-order chunk reduction -> U[h][b][:]
__global__ __launch_bounds__(256) void ured_kernel(const float* __restrict__ Upart,
                                                   float* __restrict__ U) {
  int i = blockIdx.x * 256 + threadIdx.x;            // H*B*1024 = 131072
  float s = 0.f;
  #pragma unroll
  for (int ch = 0; ch < UCH_; ++ch) s += Upart[(size_t)ch * H_ * B_ * EPROJS_ + i];
  U[i] = s;
}

// ---------------------------------------------------------------------------
// 8) c[b][h*512+e] = u[h][b] . Wv[h][:,e]
// ---------------------------------------------------------------------------
__global__ __launch_bounds__(256) void cproj_kernel(const float* __restrict__ U,
                                                    const float* __restrict__ Wv,
                                                    float* __restrict__ C) {
  int i = blockIdx.x * 256 + threadIdx.x;            // H*B*512 = 65536
  int h = i >> 14;
  int b = (i >> 9) & 31;
  int e = i & 511;
  const float* u = U + ((size_t)h * B_ + b) * EPROJS_;
  const float* w = Wv + (size_t)h * EPROJS_ * DV_ + e;
  float acc = 0.f;
  for (int d = 0; d < EPROJS_; ++d) acc += u[d] * w[(size_t)d * DV_];
  C[(size_t)b * (H_ * DV_) + h * DV_ + e] = acc;
}

// ---------------------------------------------------------------------------
// 9) out[b][j] = c[b] . Wo[:,j]
// ---------------------------------------------------------------------------
__global__ __launch_bounds__(256) void oproj_kernel(const float* __restrict__ C,
                                                    const float* __restrict__ Wo,
                                                    float* __restrict__ out) {
  int i = blockIdx.x * 256 + threadIdx.x;            // B*1024 = 32768
  int b = i >> 10;
  int j = i & 1023;
  const float* c = C + (size_t)b * (H_ * DV_);
  const float* w = Wo + j;
  float acc = 0.f;
  for (int k = 0; k < H_ * DV_; ++k) acc += c[k] * w[(size_t)k * EPROJS_];
  out[i] = acc;
}

// ---------------------------------------------------------------------------
// launch
// ---------------------------------------------------------------------------
extern "C" void kernel_launch(void* const* d_in, const int* in_sizes, int n_in,
                              void* d_out, int out_size, void* d_ws, size_t ws_size,
                              hipStream_t stream) {
  const float* enc      = (const float*)d_in[0];
  const float* dec_z    = (const float*)d_in[2];
  const float* att_prev = (const float*)d_in[3];
  const float* Wq   = (const float*)d_in[4];
  const float* bq   = (const float*)d_in[5];
  const float* Wk   = (const float*)d_in[6];
  const float* Wv   = (const float*)d_in[7];
  const float* g_w  = (const float*)d_in[8];
  const float* g_b  = (const float*)d_in[9];
  const float* Watt = (const float*)d_in[10];
  const float* Wo   = (const float*)d_in[11];
  const float* cw0  = (const float*)d_in[12];
  const float* cw1  = (const float*)d_in[13];
  const float* cw2  = (const float*)d_in[14];
  const float* cw3  = (const float*)d_in[15];

  float* out  = (float*)d_out;                 // [B, EPROJS]
  float* wout = out + B_ * EPROJS_;            // [H, B, T]

  char* ws = (char*)d_ws;                      // all offsets 256B-aligned
  const size_t XB_OFF = 0;                                        // 65,536,000
  const size_t WT_OFF = XB_OFF + (size_t)BT_ * EPROJS_ * 2;       //  5,111,808
  const size_t AP_OFF = WT_OFF + (size_t)H_ * DK_ * WTK_ * 2;     //    313,344
  const size_t QB_OFF = AP_OFF + (size_t)H_ * B_ * APL_ * 2;      //    262,144
  const size_t EB_OFF = QB_OFF + (size_t)H_ * B_ * DK_ * 4;       //    512,000
  const size_t UB_OFF = EB_OFF + (size_t)H_ * BT_ * 4;            //    524,288
  const size_t CB_OFF = UB_OFF + (size_t)H_ * B_ * EPROJS_ * 4;   //    262,144
  const size_t UP_OFF = CB_OFF + (size_t)B_ * H_ * DV_ * 4;       //  4,194,304

  unsigned short* Xb = (unsigned short*)(ws + XB_OFF);
  unsigned short* WT = (unsigned short*)(ws + WT_OFF);
  unsigned short* AP = (unsigned short*)(ws + AP_OFF);
  float* Qb = (float*)(ws + QB_OFF);
  float* Eb = (float*)(ws + EB_OFF);
  float* Ub = (float*)(ws + UB_OFF);
  float* Cb = (float*)(ws + CB_OFF);
  float* Up = (float*)(ws + UP_OFF);

  cvt_x_kernel<<<(BT_ * EPROJS_) / 1024, 256, 0, stream>>>(enc, Xb);
  build_wt_kernel<<<(H_ * DK_ * WTK_) / 256, 256, 0, stream>>>(Wk, Watt, cw0, cw1, cw2, cw3, WT);
  build_ap_kernel<<<(H_ * B_ * APL_) / 256, 256, 0, stream>>>(att_prev, AP);
  qproj_kernel<<<(H_ * B_ * DK_) / 256, 256, 0, stream>>>(dec_z, Wq, bq, Qb);

  gemm_e_kernel<<<dim3(BT_ / 32, H_), 256, 0, stream>>>(Xb, AP, WT, Qb, g_w, g_b, Eb);

  softmax_kernel<<<dim3(B_, H_), 256, 0, stream>>>(Eb, wout);
  uacc_kernel<<<dim3(B_, UCH_), 256, 0, stream>>>(enc, wout, Up);
  ured_kernel<<<(H_ * B_ * EPROJS_) / 256, 256, 0, stream>>>(Up, Ub);
  cproj_kernel<<<(H_ * B_ * DV_) / 256, 256, 0, stream>>>(Ub, Wv, Cb);
  oproj_kernel<<<(B_ * EPROJS_) / 256, 256, 0, stream>>>(Cb, Wo, out);
}
